// SINSelfAttention_63410897158176
// MI455X (gfx1250) — matrix operand order repaired and checked by reading, not compiled, not run on hardware
//
#include <hip/hip_runtime.h>

typedef __attribute__((ext_vector_type(16))) _Float16 v16h;
typedef __attribute__((ext_vector_type(8)))  float    v8f;

#define DEV __device__ __forceinline__

constexpr int kH = 64, kW = 64, kF = 64, kN = kH * kW, kB = 4;

DEV v8f wmma16(v16h a, v16h b, v8f c) {
  // D = A(16x32 f16) * B(32x16 f16) + C(16x16 f32)
  return __builtin_amdgcn_wmma_f32_16x16x32_f16(false, a, false, b, (short)0, c,
                                                false, false);
}

DEV v8f zero8() {
  v8f z;
#pragma unroll
  for (int i = 0; i < 8; ++i) z[i] = 0.f;
  return z;
}

DEV v16h zero16h() {
  v16h z;
#pragma unroll
  for (int i = 0; i < 16; ++i) z[i] = (_Float16)0.f;
  return z;
}

// A fragment: 16x32 f16, source row-major M x K (lda halves per row).
// Lane l holds row (l&15), K halves [K0,K0+8) and [K0+16,K0+24), K0 = 8*(l>>4).
DEV v16h ld_afrag(const _Float16* p, int lda, int lane) {
  const int m = lane & 15, k0 = (lane >> 4) * 8;
  v16h a;
  ((uint4*)&a)[0] = *(const uint4*)(p + (size_t)m * lda + k0);
  ((uint4*)&a)[1] = *(const uint4*)(p + (size_t)m * lda + k0 + 16);
  return a;
}

// B fragment: 32x16 f16 where B(k,n) = Bt[n][k], Bt row-major N x K (ldb halves).
// Lane l holds col n=(l&15), K halves [K0,K0+8) and [K0+16,K0+24), K0 = 8*(l>>4).
DEV v16h ld_bfrag(const _Float16* p, int ldb, int lane) {
  const int n = lane & 15, k0 = (lane >> 4) * 8;
  v16h b;
  ((uint4*)&b)[0] = *(const uint4*)(p + (size_t)n * ldb + k0);
  ((uint4*)&b)[1] = *(const uint4*)(p + (size_t)n * ldb + k0 + 16);
  return b;
}

DEV float rmax16(float x) {
#pragma unroll
  for (int d = 1; d < 16; d <<= 1) x = fmaxf(x, __shfl_xor(x, d, 16));
  return x;
}
DEV float rsum16(float x) {
#pragma unroll
  for (int d = 1; d < 16; d <<= 1) x += __shfl_xor(x, d, 16);
  return x;
}

// ---------------- prep kernels ----------------

__global__ void cvt_f16_kernel(const float* __restrict__ src,
                               _Float16* __restrict__ dst, int n) {
  int i = blockIdx.x * 256 + threadIdx.x;
  if (i < n) dst[i] = (_Float16)src[i];
}

// HWIO [3][3][ci][co] f32 -> [t][co][ci] f16 (Bt layout per tap)
__global__ void prep_w_kernel(const float* __restrict__ Wk,
                              const float* __restrict__ Wq,
                              const float* __restrict__ Wv,
                              const float* __restrict__ Wr,
                              _Float16* __restrict__ wkt,
                              _Float16* __restrict__ wqt,
                              _Float16* __restrict__ wvt,
                              _Float16* __restrict__ wrt) {
  int i = blockIdx.x * 256 + threadIdx.x;
  if (i >= 4 * 9 * 64 * 64) return;
  const int which = i / 36864;
  int r = i % 36864;
  const int t = r / 4096;
  r %= 4096;
  const int co = r / 64, ci = r % 64;
  const float* src = (which == 0) ? Wk : (which == 1) ? Wq : (which == 2) ? Wv : Wr;
  _Float16* dst = (which == 0) ? wkt : (which == 1) ? wqt : (which == 2) ? wvt : wrt;
  dst[(t * 64 + co) * 64 + ci] = (_Float16)src[(t * 64 + ci) * 64 + co];
}

// ---------------- qkv conv (implicit GEMM via WMMA) ----------------
// Each wave: 16 contiguous pixels (one row segment) x 64 channels x {k,q,v}.

__global__ __launch_bounds__(128) void conv_qkv_kernel(
    const _Float16* __restrict__ xh,
    const _Float16* __restrict__ wkt, const _Float16* __restrict__ wqt,
    const _Float16* __restrict__ wvt,
    const float* __restrict__ bk, const float* __restrict__ bq,
    const float* __restrict__ bv,
    _Float16* __restrict__ qo, _Float16* __restrict__ ko,
    _Float16* __restrict__ vto) {
  const int lane = threadIdx.x & 31;
  const int wave = threadIdx.x >> 5;
  const int b = blockIdx.x >> 6;
  const int tb = (blockIdx.x & 63) * 64 + wave * 16;  // base pixel (16-aligned => one row)
  const int h = tb >> 6, w0 = tb & 63;
  const int col = lane & 15, hi = lane >> 4;

  v8f acc[3][4];
#pragma unroll
  for (int s = 0; s < 3; ++s)
#pragma unroll
    for (int nt = 0; nt < 4; ++nt) acc[s][nt] = zero8();

  const _Float16* wt[3] = {wkt, wqt, wvt};

  for (int t = 0; t < 9; ++t) {
    const int dh = t / 3 - 1, dw = t % 3 - 1;
    const int hh = h + dh;
    const int wl = w0 + col + dw;
    const bool ok = ((unsigned)hh < 64u) && ((unsigned)wl < 64u);
    const int hc = min(max(hh, 0), 63), wc = min(max(wl, 0), 63);
    const _Float16* xrow = xh + (((size_t)(b * 64 + hc) * 64 + wc) << 6);
    const int k0 = hi * 8;
#pragma unroll
    for (int kc = 0; kc < 2; ++kc) {
      v16h a;
      if (ok) {
        ((uint4*)&a)[0] = *(const uint4*)(xrow + kc * 32 + k0);
        ((uint4*)&a)[1] = *(const uint4*)(xrow + kc * 32 + k0 + 16);
      } else {
        a = zero16h();
      }
#pragma unroll
      for (int s = 0; s < 3; ++s) {
        const _Float16* wb = wt[s] + t * 4096 + kc * 32;
#pragma unroll
        for (int nt = 0; nt < 4; ++nt)
          acc[s][nt] = wmma16(a, ld_bfrag(wb + nt * 16 * 64, 64, lane), acc[s][nt]);
      }
    }
  }

#pragma unroll
  for (int s = 0; s < 3; ++s) {
    const float* bias = (s == 0) ? bk : (s == 1) ? bq : bv;
#pragma unroll
    for (int nt = 0; nt < 4; ++nt) {
      const int co = nt * 16 + col;
      const float bb = bias[co];
#pragma unroll
      for (int r = 0; r < 8; ++r) {
        float y = acc[s][nt][r] + bb;
        const int p = tb + r + 8 * hi;
        if (s == 2) {  // v: relu, stored transposed [F][N]
          y = fmaxf(y, 0.f);
          vto[((size_t)b * 64 + co) * kN + p] = (_Float16)y;
        } else {       // k,q: elu, stored [N][F]
          y = (y > 0.f) ? y : (__expf(y) - 1.f);
          _Float16* dst = (s == 0) ? ko : qo;
          dst[((size_t)b * kN + p) * 64 + co] = (_Float16)y;
        }
      }
    }
  }
}

// ---------------- flash attention ----------------
// Each wave: 16 query rows, online softmax over all 4096 keys in 64-key blocks
// (4 score sub-tiles per block to amortize cross-lane reductions).
// Diagonal score is zeroed (not -inf) per the reference, then scaled by 1/8.

__global__ __launch_bounds__(128) void attn_kernel(
    const _Float16* __restrict__ q, const _Float16* __restrict__ k,
    const _Float16* __restrict__ vt, _Float16* __restrict__ ah) {
  __shared__ alignas(16) _Float16 pstage[4][16][72];  // per-wave P-tile staging
  const int lane = threadIdx.x & 31, wave = threadIdx.x >> 5;
  const int b = blockIdx.x >> 6;
  const int qbase = (blockIdx.x & 63) * 64 + wave * 16;
  const int col = lane & 15, hi = lane >> 4;

  const _Float16* qb = q + (size_t)b * kN * 64;
  const _Float16* kb = k + (size_t)b * kN * 64;
  const _Float16* vb = vt + (size_t)b * 64 * kN;

  const v16h aq0 = ld_afrag(qb + (size_t)qbase * 64, 64, lane);
  const v16h aq1 = ld_afrag(qb + (size_t)qbase * 64 + 32, 64, lane);

  v8f acc[4];
#pragma unroll
  for (int ft = 0; ft < 4; ++ft) acc[ft] = zero8();
  float mrow[8], lrow[8];
#pragma unroll
  for (int r = 0; r < 8; ++r) { mrow[r] = -1e30f; lrow[r] = 0.f; }

  _Float16* ps = &pstage[wave][0][0];

  for (int j = 0; j < kN; j += 64) {
    // S = Q * K^T: four 16-key sub-tiles, K=64 via two chained WMMAs each
    v8f s[4];
#pragma unroll
    for (int u = 0; u < 4; ++u) {
      s[u] = zero8();
      s[u] = wmma16(aq0, ld_bfrag(kb + (size_t)(j + 16 * u) * 64, 64, lane), s[u]);
      s[u] = wmma16(aq1, ld_bfrag(kb + (size_t)(j + 16 * u) * 64 + 32, 64, lane),
                    s[u]);
    }

    if (j + 64 < kN)
      __builtin_prefetch((const void*)(kb + (size_t)(j + 64) * 64), 0, 1);

    const int bcol = j + col;  // key index of sub-tile 0 in this lane's column
#pragma unroll
    for (int r = 0; r < 8; ++r) {
      const int row = qbase + r + 8 * hi;
      const int d = row - bcol;  // diagonal hits at d == 16*u
      float e[4];
#pragma unroll
      for (int u = 0; u < 4; ++u) {
        e[u] = s[u][r] * 0.125f;
        if (d == 16 * u) e[u] = 0.f;  // diagonal zeroed, still in softmax
      }
      float mx = fmaxf(fmaxf(e[0], e[1]), fmaxf(e[2], e[3]));
      mx = rmax16(mx);
      const float nm = fmaxf(mrow[r], mx);
      const float corr = __expf(mrow[r] - nm);
      float p[4], psum = 0.f;
#pragma unroll
      for (int u = 0; u < 4; ++u) {
        p[u] = __expf(e[u] - nm);
        psum += p[u];
      }
      lrow[r] = lrow[r] * corr + rsum16(psum);
      mrow[r] = nm;
#pragma unroll
      for (int ft = 0; ft < 4; ++ft) acc[ft][r] *= corr;
      _Float16* prow = ps + (r + 8 * hi) * 72 + col;
#pragma unroll
      for (int u = 0; u < 4; ++u) prow[16 * u] = (_Float16)p[u];  // C-layout -> LDS
    }
    // reload P (16x64) as two A fragments (in-wave DS ordering; dscnt waits)
    const v16h pa0 = ld_afrag(ps, 72, lane);
    const v16h pa1 = ld_afrag(ps + 32, 72, lane);
#pragma unroll
    for (int ft = 0; ft < 4; ++ft) {
      acc[ft] = wmma16(pa0, ld_bfrag(vb + (size_t)(ft * 16) * kN + j, kN, lane),
                       acc[ft]);
      acc[ft] = wmma16(
          pa1, ld_bfrag(vb + (size_t)(ft * 16) * kN + j + 32, kN, lane), acc[ft]);
    }
  }

#pragma unroll
  for (int ft = 0; ft < 4; ++ft)
#pragma unroll
    for (int r = 0; r < 8; ++r) {
      const float o = acc[ft][r] / lrow[r];
      ah[((size_t)b * kN + qbase + r + 8 * hi) * 64 + ft * 16 + col] = (_Float16)o;
    }
}

// ---------------- output conv ----------------

__global__ __launch_bounds__(128) void conv_out_kernel(
    const _Float16* __restrict__ ahh, const _Float16* __restrict__ wrt,
    const float* __restrict__ br, float* __restrict__ out) {
  const int lane = threadIdx.x & 31;
  const int wave = threadIdx.x >> 5;
  const int b = blockIdx.x >> 6;
  const int tb = (blockIdx.x & 63) * 64 + wave * 16;
  const int h = tb >> 6, w0 = tb & 63;
  const int col = lane & 15, hi = lane >> 4;

  v8f acc[4];
#pragma unroll
  for (int nt = 0; nt < 4; ++nt) acc[nt] = zero8();

  for (int t = 0; t < 9; ++t) {
    const int dh = t / 3 - 1, dw = t % 3 - 1;
    const int hh = h + dh;
    const int wl = w0 + col + dw;
    const bool ok = ((unsigned)hh < 64u) && ((unsigned)wl < 64u);
    const int hc = min(max(hh, 0), 63), wc = min(max(wl, 0), 63);
    const _Float16* xrow = ahh + (((size_t)(b * 64 + hc) * 64 + wc) << 6);
    const int k0 = hi * 8;
#pragma unroll
    for (int kc = 0; kc < 2; ++kc) {
      v16h a;
      if (ok) {
        ((uint4*)&a)[0] = *(const uint4*)(xrow + kc * 32 + k0);
        ((uint4*)&a)[1] = *(const uint4*)(xrow + kc * 32 + k0 + 16);
      } else {
        a = zero16h();
      }
      const _Float16* wb = wrt + t * 4096 + kc * 32;
#pragma unroll
      for (int nt = 0; nt < 4; ++nt)
        acc[nt] = wmma16(a, ld_bfrag(wb + nt * 16 * 64, 64, lane), acc[nt]);
    }
  }

#pragma unroll
  for (int nt = 0; nt < 4; ++nt) {
    const int co = nt * 16 + col;
    const float bb = br[co];
#pragma unroll
    for (int r = 0; r < 8; ++r) {
      const int p = tb + r + 8 * hi;
      out[((size_t)b * kN + p) * 64 + co] = fmaxf(acc[nt][r] + bb, 0.f);
    }
  }
}

// ---------------- launch ----------------

extern "C" void kernel_launch(void* const* d_in, const int* in_sizes, int n_in,
                              void* d_out, int out_size, void* d_ws,
                              size_t ws_size, hipStream_t stream) {
  (void)in_sizes; (void)n_in; (void)out_size; (void)ws_size;
  const float* x  = (const float*)d_in[0];
  const float* Wk = (const float*)d_in[1];
  const float* bk = (const float*)d_in[2];
  const float* Wq = (const float*)d_in[3];
  const float* bq = (const float*)d_in[4];
  const float* Wv = (const float*)d_in[5];
  const float* bv = (const float*)d_in[6];
  const float* Wr = (const float*)d_in[7];
  const float* br = (const float*)d_in[8];

  char* ws = (char*)d_ws;
  const size_t MB = 1u << 20;
  _Float16* xh  = (_Float16*)(ws + 0 * MB);   // [B][N][64] f16   (2 MB)
  _Float16* qh  = (_Float16*)(ws + 2 * MB);   // [B][N][64] f16
  _Float16* kh  = (_Float16*)(ws + 4 * MB);   // [B][N][64] f16
  _Float16* vth = (_Float16*)(ws + 6 * MB);   // [B][64][N] f16 (transposed)
  _Float16* ahh = (_Float16*)(ws + 8 * MB);   // [B][N][64] f16 (attention out)
  _Float16* wkt = (_Float16*)(ws + 10 * MB);  // 4x [9][64][64] f16
  _Float16* wqt = wkt + 9 * 64 * 64;
  _Float16* wvt = wqt + 9 * 64 * 64;
  _Float16* wrt = wvt + 9 * 64 * 64;

  const int nx = kB * kN * kF;  // 1,048,576
  cvt_f16_kernel<<<nx / 256, 256, 0, stream>>>(x, xh, nx);
  prep_w_kernel<<<(4 * 9 * 64 * 64) / 256, 256, 0, stream>>>(
      Wk, Wq, Wv, Wr, wkt, wqt, wvt, wrt);
  conv_qkv_kernel<<<kB * 64, 128, 0, stream>>>(xh, wkt, wqt, wvt, bk, bq, bv,
                                               qh, kh, vth);
  attn_kernel<<<kB * 64, 128, 0, stream>>>(qh, kh, vth, ahh);
  conv_out_kernel<<<kB * 64, 128, 0, stream>>>(ahh, wrt, br, (float*)d_out);
}
